// DiffAttn_17849884082658
// MI455X (gfx1250) — compile-verified, matrix operand order checked by reading, not run on hardware
//
#include <hip/hip_runtime.h>
#include <hip/hip_bf16.h>

// ---------------------------------------------------------------------------
// DiffAttn on gfx1250: bf16 WMMA everywhere, flash-style dual softmax,
// async global->LDS staging of the Q tile.
// ---------------------------------------------------------------------------

typedef __attribute__((ext_vector_type(16))) __bf16          v16bf;
typedef __attribute__((ext_vector_type(8)))  float           v8f;
typedef __attribute__((ext_vector_type(8)))  unsigned short  u16x8;
typedef int v4i __attribute__((vector_size(16)));   // matches builtin param type

constexpr int kB = 4;
constexpr int kS = 2048;
constexpr int kD = 1024;
constexpr int kH = 512;           // half head dim
constexpr int kQPad = 1032;       // LDS row stride (shorts): 516 dwords -> +4 banks/row

#define AS1 __attribute__((address_space(1)))
#define AS3 __attribute__((address_space(3)))

#if __has_builtin(__builtin_amdgcn_global_load_async_to_lds_b128)
#define HAVE_ASYNC_LDS 1
#endif

// fp32 -> bf16 round-to-nearest-even
__device__ __forceinline__ unsigned short f2bf(float f) {
    unsigned u = __float_as_uint(f);
    u += 0x7FFFu + ((u >> 16) & 1u);
    return (unsigned short)(u >> 16);
}

__device__ __forceinline__ v8f zero8() {
    v8f z;
#pragma unroll
    for (int i = 0; i < 8; ++i) z[i] = 0.0f;
    return z;
}

// Build a v16bf A/B fragment from two contiguous 8-element (16B) chunks.
// Per-lane ISA layout (16-bit 16x32): elems 0..7 -> K = hi*8 + e,
// elems 8..15 -> K = hi*8 + 16 + (e-8). Caller passes the two chunk offsets.
__device__ __forceinline__ v16bf load_frag(const unsigned short* __restrict__ base,
                                           int o1, int o2) {
    union { v16bf v; u16x8 h[2]; } f;
    f.h[0] = *reinterpret_cast<const u16x8*>(base + o1);
    f.h[1] = *reinterpret_cast<const u16x8*>(base + o2);
    return f.v;
}

__device__ __forceinline__ v8f wmma_bf16(v16bf a, v16bf b, v8f c) {
    return __builtin_amdgcn_wmma_f32_16x16x32_bf16(
        /*neg_a=*/false, a, /*neg_b=*/false, b,
        /*c_mod=*/(short)0, c, /*reuse_a=*/false, /*reuse_b=*/false);
}

// 16-byte async global->LDS copy (ASYNCcnt-tracked); fallback: direct copy.
__device__ __forceinline__ void async_copy16(const unsigned short* g, unsigned short* l) {
#if defined(HAVE_ASYNC_LDS)
    __builtin_amdgcn_global_load_async_to_lds_b128(
        (AS1 v4i*)(unsigned long long)g,
        (AS3 v4i*)(unsigned)(unsigned long long)l,
        /*offset=*/0, /*cpol=*/0);
#else
    *reinterpret_cast<u16x8*>(l) = *reinterpret_cast<const u16x8*>(g);
#endif
}

__device__ __forceinline__ void async_wait0() {
#if defined(HAVE_ASYNC_LDS)
#if __has_builtin(__builtin_amdgcn_s_wait_asynccnt)
    __builtin_amdgcn_s_wait_asynccnt(0);
#else
    asm volatile("s_wait_asynccnt 0" ::: "memory");
#endif
#endif
}

// ---------------------------------------------------------------------------
// Kernel 1: fp32 -> bf16 cast (grid-stride)
// ---------------------------------------------------------------------------
__global__ void cast_f32_to_bf16(const float* __restrict__ in,
                                 unsigned short* __restrict__ out, int n) {
    int i = blockIdx.x * blockDim.x + threadIdx.x;
    int stride = gridDim.x * blockDim.x;
    for (; i < n; i += stride) out[i] = f2bf(in[i]);
}

// ---------------------------------------------------------------------------
// Kernel 2: QKV projection. C[m,n] = sum_k hs[m,k]*W[n,k] + bias[n].
// 8 waves/block, each wave one 16x16 tile (block tile 32x64). z = {Q,K,V}.
// V is written transposed: Vt[b][d][s], so attn@V B-frags are contiguous.
// ---------------------------------------------------------------------------
__global__ __launch_bounds__(256)
void qkv_gemm(const unsigned short* __restrict__ hs,   // [B*S, D] bf16
              const unsigned short* __restrict__ wAll, // [3, D(n), D(k)] bf16
              const float* __restrict__ bq,
              const float* __restrict__ bk,
              const float* __restrict__ bvp,
              unsigned short* __restrict__ Qb,         // [B*S, D] bf16
              unsigned short* __restrict__ Kb,         // [B*S, D] bf16
              unsigned short* __restrict__ Vt) {       // [B, D, S] bf16
    const int lane = threadIdx.x & 31;
    const int wid  = threadIdx.x >> 5;
    const int hi   = lane >> 4;
    const int ln   = lane & 15;

    const int m0  = blockIdx.x * 32 + (wid >> 2) * 16;
    const int n0  = blockIdx.y * 64 + (wid & 3) * 16;
    const int mat = blockIdx.z;

    const unsigned short* W    = wAll + (size_t)mat * kD * kD;
    const float*          bias = (mat == 0) ? bq : ((mat == 1) ? bk : bvp);

    // A: row of hidden states; B: row n of W (B[k,n] = W[n,k] -> contiguous k)
    const unsigned short* arow = hs + (size_t)(m0 + ln) * kD;
    const unsigned short* brow = W  + (size_t)(n0 + ln) * kD;

    v8f acc = zero8();
#pragma unroll 4
    for (int k0 = 0; k0 < kD; k0 += 32) {
        v16bf a = load_frag(arow, k0 + hi * 8, k0 + hi * 8 + 16);
        v16bf b = load_frag(brow, k0 + hi * 8, k0 + hi * 8 + 16);
        acc = wmma_bf16(a, b, acc);
    }

    const float bval = bias[n0 + ln];
    if (mat < 2) {
        unsigned short* outp = (mat == 0) ? Qb : Kb;
#pragma unroll
        for (int r = 0; r < 8; ++r) {
            const int row = m0 + r + hi * 8;
            outp[(size_t)row * kD + n0 + ln] = f2bf(acc[r] + bval);
        }
    } else {
#pragma unroll
        for (int r = 0; r < 8; ++r) {
            const int row = m0 + r + hi * 8;
            const int bb  = row >> 11;        // / 2048
            const int ss  = row & (kS - 1);   // % 2048
            Vt[((size_t)bb * kD + n0 + ln) * kS + ss] = f2bf(acc[r] + bval);
        }
    }
}

// ---------------------------------------------------------------------------
// Kernel 3: flash-style differential attention.
// Block = (batch, 16-row q tile), 256 threads = 8 waves.
// Q tile (16x1024 bf16) staged once into LDS via async global->LDS loads.
// Phase 1: waves 0..7 -> score block (half hf = wid>>2, key sub-col wid&3).
// Phase 2: dual online softmax + two WMMA accumulators; wave owns 128 out cols.
// ---------------------------------------------------------------------------
__global__ __launch_bounds__(256)
void diff_attn(const unsigned short* __restrict__ Qb,
               const unsigned short* __restrict__ Kb,
               const unsigned short* __restrict__ Vt,
               const float* __restrict__ scalarp,
               float* __restrict__ out) {
    __shared__ __align__(16) unsigned short q_lds[16][kQPad]; // padded Q tile
    __shared__ float sc_lds[2][16][64];                       // scaled scores

    const int lane = threadIdx.x & 31;
    const int wid  = threadIdx.x >> 5;
    const int hi   = lane >> 4;
    const int ln   = lane & 15;
    const int tid  = threadIdx.x;

    const int qt = blockIdx.x & (kS / 16 - 1);
    const int b  = blockIdx.x >> 7;
    const int q0 = qt * 16;

    const float scl  = scalarp[0];
    const float smax = 0.03125f;          // 1/sqrt(1024)

    // ---- stage Q tile: 16 rows x 1024 cols bf16 = 2048 x 16B chunks -------
    {
        const unsigned short* qbase = Qb + ((size_t)b * kS + q0) * kD;
#pragma unroll
        for (int i = 0; i < 8; ++i) {
            const int c   = tid + 256 * i;    // chunk id 0..2047
            const int row = c >> 7;           // 128 chunks per row
            const int col = (c & 127) * 8;    // element offset
            async_copy16(qbase + (size_t)row * kD + col, &q_lds[row][col]);
        }
        async_wait0();
    }
    __syncthreads();

    const int hf = wid >> 2;              // which half of head dim for scores
    const int cb = wid & 3;               // key sub-column (16 keys)
    const unsigned short* qrow = &q_lds[ln][hf * kH];

    float m1 = -3.0e38f, l1 = 0.0f, m2 = -3.0e38f, l2 = 0.0f;
    v8f o1[8], o2[8];
#pragma unroll
    for (int j = 0; j < 8; ++j) { o1[j] = zero8(); o2[j] = zero8(); }

    for (int n0 = 0; n0 < kS; n0 += 64) {
        // -------- phase 1: 16x16 score block via 16 chained WMMAs ----------
        const unsigned short* krow =
            Kb + ((size_t)b * kS + n0 + cb * 16 + ln) * kD + hf * kH;
        if (n0 + 64 < kS)
            __builtin_prefetch(Kb + ((size_t)b * kS + n0 + 64 + cb * 16 + ln) * kD + hf * kH, 0, 1);

        v8f sc = zero8();
#pragma unroll 4
        for (int k0 = 0; k0 < kH; k0 += 32) {
            v16bf a  = load_frag(qrow, k0 + hi * 8, k0 + hi * 8 + 16); // LDS
            v16bf bb = load_frag(krow, k0 + hi * 8, k0 + hi * 8 + 16); // global
            sc = wmma_bf16(a, bb, sc);
        }
#pragma unroll
        for (int r = 0; r < 8; ++r)
            sc_lds[hf][r + hi * 8][cb * 16 + ln] = sc[r] * smax;

        __syncthreads();

        // -------- phase 2: dual online softmax (per-wave, row = ln) --------
        float tm1 = -3.0e38f, tm2 = -3.0e38f;
#pragma unroll 8
        for (int c = 0; c < 64; ++c) {
            tm1 = fmaxf(tm1, sc_lds[0][ln][c]);
            tm2 = fmaxf(tm2, sc_lds[1][ln][c]);
        }
        const float nm1 = fmaxf(m1, tm1), nm2 = fmaxf(m2, tm2);
        const float al1 = __expf(m1 - nm1), al2 = __expf(m2 - nm2);
        float s1 = 0.0f, s2 = 0.0f;
#pragma unroll 8
        for (int c = 0; c < 64; ++c) {
            s1 += __expf(sc_lds[0][ln][c] - nm1);
            s2 += __expf(sc_lds[1][ln][c] - nm2);
        }
        l1 = l1 * al1 + s1; l2 = l2 * al2 + s2; m1 = nm1; m2 = nm2;

        // rescale accumulators (broadcast per-row alpha via shuffle)
        float a1r[8], a2r[8];
#pragma unroll
        for (int r = 0; r < 8; ++r) {
            a1r[r] = __shfl(al1, r + hi * 8, 32);
            a2r[r] = __shfl(al2, r + hi * 8, 32);
        }
#pragma unroll
        for (int j = 0; j < 8; ++j)
#pragma unroll
            for (int r = 0; r < 8; ++r) { o1[j][r] *= a1r[r]; o2[j][r] *= a2r[r]; }

        // P tiles in A-fragment layout straight from LDS (transpose via LDS)
#pragma unroll
        for (int t = 0; t < 2; ++t) {
            union { v16bf v; unsigned short s[16]; } p1, p2;
#pragma unroll
            for (int e = 0; e < 16; ++e) {
                const int kl = 32 * t + hi * 8 + e + (e >= 8 ? 8 : 0);
                p1.s[e] = f2bf(__expf(sc_lds[0][ln][kl] - nm1));
                p2.s[e] = f2bf(__expf(sc_lds[1][ln][kl] - nm2));
            }
#pragma unroll
            for (int j = 0; j < 8; ++j) {
                // B[k,n] = V[key, d] = Vt[d][key]: contiguous in key
                const unsigned short* vrow =
                    Vt + ((size_t)b * kD + wid * 128 + j * 16 + ln) * kS + n0 + 32 * t;
                v16bf vb = load_frag(vrow, hi * 8, hi * 8 + 16);
                o1[j] = wmma_bf16(p1.v, vb, o1[j]);
                o2[j] = wmma_bf16(p2.v, vb, o2[j]);
            }
        }
        __syncthreads();
    }

    // final combine: O1/l1 - scalar * O2/l2
    const float inv1 = 1.0f / l1;
    const float inv2 = scl / l2;
    float r1[8], r2[8];
#pragma unroll
    for (int r = 0; r < 8; ++r) {
        r1[r] = __shfl(inv1, r + hi * 8, 32);
        r2[r] = __shfl(inv2, r + hi * 8, 32);
    }
#pragma unroll
    for (int j = 0; j < 8; ++j)
#pragma unroll
        for (int r = 0; r < 8; ++r) {
            const size_t idx = ((size_t)b * kS + q0 + r + hi * 8) * (size_t)kD
                             + wid * 128 + j * 16 + ln;
            out[idx] = o1[j][r] * r1[r] - o2[j][r] * r2[r];
        }
}

// ---------------------------------------------------------------------------
// Launch
// ---------------------------------------------------------------------------
extern "C" void kernel_launch(void* const* d_in, const int* in_sizes, int n_in,
                              void* d_out, int out_size, void* d_ws, size_t ws_size,
                              hipStream_t stream) {
    const float* hs  = (const float*)d_in[0];
    const float* Wq  = (const float*)d_in[1];
    const float* bq  = (const float*)d_in[2];
    const float* Wk  = (const float*)d_in[3];
    const float* bk  = (const float*)d_in[4];
    const float* Wv  = (const float*)d_in[5];
    const float* bvp = (const float*)d_in[6];
    const float* scl = (const float*)d_in[7];
    float* out = (float*)d_out;

    const size_t nHS = (size_t)kB * kS * kD;   // 8,388,608
    const int    nW  = kD * kD;                // 1,048,576

    unsigned short* hs_b = (unsigned short*)d_ws;
    unsigned short* w_b  = hs_b + nHS;            // 3 * D*D
    unsigned short* q_b  = w_b + (size_t)3 * nW;
    unsigned short* k_b  = q_b + nHS;
    unsigned short* vt_b = k_b + nHS;             // total ~73.4 MB

    cast_f32_to_bf16<<<2048, 256, 0, stream>>>(hs, hs_b, (int)nHS);
    cast_f32_to_bf16<<<1024, 256, 0, stream>>>(Wq, w_b,           nW);
    cast_f32_to_bf16<<<1024, 256, 0, stream>>>(Wk, w_b + nW,      nW);
    cast_f32_to_bf16<<<1024, 256, 0, stream>>>(Wv, w_b + 2 * nW,  nW);

    dim3 ggrid((kB * kS) / 32, kD / 64, 3);
    qkv_gemm<<<ggrid, 256, 0, stream>>>(hs_b, w_b, bq, bk, bvp, q_b, k_b, vt_b);

    diff_attn<<<kB * (kS / 16), 256, 0, stream>>>(q_b, k_b, vt_b, scl, out);
}
